// Mamba2Block_17111149707385
// MI455X (gfx1250) — matrix operand order repaired and checked by reading, not compiled, not run on hardware
//
#include <hip/hip_runtime.h>
#include <hip/hip_bf16.h>
#include <stdint.h>

#define DMODEL  1024
#define DSTATE  64
#define HEADDIM 64
#define DINNER  2048
#define NHEADS  32
#define CONVDIM 2176
#define DINPROJ 4256
#define BSZ     4
#define SEQLEN  2048
#define NCH     32
#define NTOK    (BSZ*SEQLEN)
#define EPSF    1e-5f

typedef __attribute__((ext_vector_type(16))) __bf16 v16bf;
typedef __attribute__((ext_vector_type(8)))  float  v8f;

union FragB { v16bf v; uint32_t u[8]; uint4 q[2]; };

__device__ __forceinline__ uint16_t f2bf(float f) {
  union { float f; uint32_t u; } c; c.f = f;
  uint32_t u = c.u;
  return (uint16_t)((u + 0x7FFFu + ((u >> 16) & 1u)) >> 16);  // RNE
}
__device__ __forceinline__ uint32_t pk2(float a, float b) {
  return (uint32_t)f2bf(a) | ((uint32_t)f2bf(b) << 16);
}
__device__ __forceinline__ v8f wmma_bf(const FragB& a, const FragB& b, v8f c) {
  return __builtin_amdgcn_wmma_f32_16x16x32_bf16(false, a.v, false, b.v,
                                                 (short)0, c, false, false);
}
__device__ __forceinline__ float sigm(float x) { return 1.0f / (1.0f + expf(-x)); }

// ------------------- one-shot f32 -> packed bf16 conversion ------------------
__global__ void __launch_bounds__(256)
cvt_bf16x4(const float* __restrict__ src, uint16_t* __restrict__ dst, int n4)
{
  int i = blockIdx.x * blockDim.x + threadIdx.x;
  if (i >= n4) return;
  const float4 v = ((const float4*)src)[i];
  uint2 o; o.x = pk2(v.x, v.y); o.y = pk2(v.z, v.w);
  ((uint2*)dst)[i] = o;
}

// ---------------- big NT GEMM, both operands packed bf16 ---------------------
// C[M,N] = A[M,K] * B[N,K]^T ; block tile 128x64, wave tile 64x32 (8 WMMAs/k)
__global__ void __launch_bounds__(128)
gemm_nt_bb(const uint16_t* __restrict__ A, const uint16_t* __restrict__ B,
           float* __restrict__ C, int M, int N, int K)
{
  (void)M;
  const int lane = threadIdx.x & 31, wave = threadIdx.x >> 5;
  const int mr = lane & 15, hi = lane >> 4;
  const int wRow = blockIdx.y * 128 + (wave >> 1) * 64;
  const int wCol = blockIdx.x * 64 + (wave & 1) * 32;

  v8f acc[4][2] = {};
  for (int kk = 0; kk < K; kk += 32) {
    FragB af[4], bfr[2];
#pragma unroll
    for (int i = 0; i < 4; ++i) {
      const size_t rbase = (size_t)(wRow + i * 16 + mr) * K + kk;
      af[i].q[0] = *(const uint4*)(A + rbase + hi * 8);
      af[i].q[1] = *(const uint4*)(A + rbase + 16 + hi * 8);
    }
#pragma unroll
    for (int j = 0; j < 2; ++j) {
      int n = wCol + j * 16 + mr; if (n >= N) n = N - 1;
      const uint4* pb = (const uint4*)(B + (size_t)n * K + kk + hi * 8);
      bfr[j].q[0] = pb[0];
      bfr[j].q[1] = pb[2];
    }
#pragma unroll
    for (int i = 0; i < 4; ++i)
#pragma unroll
      for (int j = 0; j < 2; ++j)
        acc[i][j] = wmma_bf(af[i], bfr[j], acc[i][j]);
  }
#pragma unroll
  for (int i = 0; i < 4; ++i) {
    int rbase = wRow + i * 16 + hi * 8;
#pragma unroll
    for (int j = 0; j < 2; ++j) {
      int col = wCol + j * 16 + mr;
      if (col < N) {
#pragma unroll
        for (int r = 0; r < 8; ++r)
          C[(size_t)(rbase + r) * N + col] = acc[i][j][r];
      }
    }
  }
}

// ---------------------------- dt = softplus ----------------------------------
__global__ void dt_softplus(const float* __restrict__ zxbcdt,
                            const float* __restrict__ dt_bias,
                            float* __restrict__ dt)
{
  int idx = blockIdx.x * blockDim.x + threadIdx.x;
  if (idx >= NTOK * NHEADS) return;
  int tok = idx >> 5, h = idx & 31;
  float v = zxbcdt[(size_t)tok * DINPROJ + (DINNER + CONVDIM) + h] + dt_bias[h];
  dt[idx] = (v > 20.f) ? v : log1pf(expf(v));
}

// ----------------------- depthwise causal conv + SiLU ------------------------
__global__ void __launch_bounds__(256)
conv_silu(const float* __restrict__ zxbcdt, const float* __restrict__ conv_w,
          const float* __restrict__ conv_b, float* __restrict__ xq,
          float* __restrict__ Bm, float* __restrict__ Cm)
{
  const int tok = blockIdx.x;
  const int t = tok & (SEQLEN - 1);
  for (int ch = threadIdx.x; ch < CONVDIM; ch += blockDim.x) {
    float acc = conv_b[ch];
    const float* w = conv_w + ch * 4;
#pragma unroll
    for (int j = 0; j < 4; ++j) {
      int tt = t - 3 + j;
      if (tt >= 0) acc += w[j] * zxbcdt[(size_t)(tok - 3 + j) * DINPROJ + DINNER + ch];
    }
    float v = acc * sigm(acc);
    if (ch < DINNER)               xq[(size_t)tok * DINNER + ch] = v;
    else if (ch < DINNER + DSTATE) Bm[(size_t)tok * DSTATE + (ch - DINNER)] = v;
    else                           Cm[(size_t)tok * DSTATE + (ch - DINNER - DSTATE)] = v;
  }
}

// --------- 64x64x64 bf16 WMMA GEMM from LDS (b128 fragment loads) ------------
__device__ __forceinline__ void gemm64(const uint16_t* __restrict__ Ash,
                                       const uint16_t* __restrict__ Bsh,
                                       int lane, int wave, v8f acc[2][2])
{
  const int mr = lane & 15, hi = lane >> 4;
  const int rowB = (wave >> 1) * 32, colB = (wave & 1) * 32;
#pragma unroll
  for (int kk = 0; kk < 64; kk += 32) {
    FragB af[2], bfr[2];
#pragma unroll
    for (int i = 0; i < 2; ++i) {
      const uint16_t* pa = Ash + (rowB + i * 16 + mr) * 64 + kk + hi * 8;
      af[i].q[0] = *(const uint4*)(pa);
      af[i].q[1] = *(const uint4*)(pa + 16);
    }
#pragma unroll
    for (int j = 0; j < 2; ++j) {
      const uint4* pb = (const uint4*)(Bsh + (colB + j * 16 + mr) * 64 + kk + hi * 8);
      bfr[j].q[0] = pb[0];
      bfr[j].q[1] = pb[2];
    }
#pragma unroll
    for (int i = 0; i < 2; ++i)
#pragma unroll
      for (int j = 0; j < 2; ++j)
        acc[i][j] = wmma_bf(af[i], bfr[j], acc[i][j]);
  }
}

// ------------- per (b,chunk,head): Y_diag + local chunk states ---------------
__global__ void __launch_bounds__(128)
ssm_chunk(const float* __restrict__ xq, const float* __restrict__ Bm,
          const float* __restrict__ Cm, const float* __restrict__ dt,
          const float* __restrict__ A_log,
          float* __restrict__ y, float* __restrict__ statesL,
          float* __restrict__ Acumlast)
{
  const int h = blockIdx.x, c = blockIdx.y, b = blockIdx.z;
  const int tid = threadIdx.x, lane = tid & 31, wave = tid >> 5;
  __shared__ float sdt[64], sAc[64];
  __shared__ __align__(16) uint16_t Cl[64 * 64], Bl[64 * 64], BT[64 * 64],
                                    XT[64 * 64], XDT[64 * 64], Ml[64 * 64];
  const int tok0 = b * SEQLEN + c * 64;

  if (tid < 64) sdt[tid] = dt[(size_t)(tok0 + tid) * NHEADS + h];
  __syncthreads();
  if (tid == 0) {
    float Aval = -expf(A_log[h]), run = 0.f;
    for (int l = 0; l < 64; ++l) { run += Aval * sdt[l]; sAc[l] = run; }
    Acumlast[((size_t)(b * NCH + c)) * NHEADS + h] = run;
  }
  __syncthreads();
  const float acl = sAc[63];
  for (int idx = tid; idx < 4096; idx += 128) {
    int l = idx >> 6, n = idx & 63;
    size_t tokl = (size_t)(tok0 + l);
    float cv = Cm[tokl * DSTATE + n];
    float bv = Bm[tokl * DSTATE + n];
    Cl[l * 64 + n] = f2bf(cv);
    Bl[l * 64 + n] = f2bf(bv);
    BT[n * 64 + l] = f2bf(bv);
    float xs = xq[tokl * DINNER + h * HEADDIM + n] * sdt[l];   // p == n
    XT[n * 64 + l]  = f2bf(xs);
    XDT[n * 64 + l] = f2bf(xs * expf(acl - sAc[l]));
  }
  __syncthreads();

  const int rowB = (wave >> 1) * 32, colB = (wave & 1) * 32;
  const int mr = lane & 15, hi = lane >> 4;

  { // GEMM1: CB^T, then apply causal decay mask L -> Ml (bf16)
    v8f acc[2][2] = {};
    gemm64(Cl, Bl, lane, wave, acc);
#pragma unroll
    for (int i = 0; i < 2; ++i) {
      int rb = rowB + i * 16 + hi * 8;
#pragma unroll
      for (int j = 0; j < 2; ++j) {
        int cc = colB + j * 16 + mr;
#pragma unroll
        for (int r = 0; r < 8; ++r) {
          int rr = rb + r;
          float v = (rr >= cc) ? acc[i][j][r] * expf(sAc[rr] - sAc[cc]) : 0.f;
          Ml[rr * 64 + cc] = f2bf(v);
        }
      }
    }
  }
  __syncthreads();
  { // GEMM2: Y_diag = M @ xs
    v8f acc[2][2] = {};
    gemm64(Ml, XT, lane, wave, acc);
#pragma unroll
    for (int i = 0; i < 2; ++i) {
      int rb = rowB + i * 16 + hi * 8;
#pragma unroll
      for (int j = 0; j < 2; ++j) {
        int pp = colB + j * 16 + mr;
#pragma unroll
        for (int r = 0; r < 8; ++r)
          y[((size_t)(tok0 + rb + r)) * DINNER + h * HEADDIM + pp] = acc[i][j][r];
      }
    }
  }
  { // GEMM3: local states[p,n] = sum_l decay(l)*xs[l,p]*B[l,n]
    v8f acc[2][2] = {};
    gemm64(XDT, BT, lane, wave, acc);
    size_t base = ((size_t)((b * NCH + c) * NHEADS + h)) * 4096;
#pragma unroll
    for (int i = 0; i < 2; ++i) {
      int rb = rowB + i * 16 + hi * 8;
#pragma unroll
      for (int j = 0; j < 2; ++j) {
        int nn = colB + j * 16 + mr;
#pragma unroll
        for (int r = 0; r < 8; ++r)
          statesL[base + (size_t)(rb + r) * 64 + nn] = acc[i][j][r];
      }
    }
  }
}

// ------------------- inter-chunk state recurrence (scan) ---------------------
__global__ void __launch_bounds__(256)
state_scan(const float* __restrict__ statesL, const float* __restrict__ Acumlast,
           float* __restrict__ statesIn)
{
  const int h = blockIdx.x, b = blockIdx.y, tid = threadIdx.x;
  float S[16];
#pragma unroll
  for (int j = 0; j < 16; ++j) S[j] = 0.f;
  for (int c = 0; c < NCH; ++c) {
    size_t base = ((size_t)((b * NCH + c) * NHEADS + h)) * 4096;
    float sc = expf(Acumlast[(size_t)(b * NCH + c) * NHEADS + h]);
#pragma unroll
    for (int j = 0; j < 16; ++j) {
      size_t e = base + tid + j * 256;
      statesIn[e] = S[j];
      S[j] = S[j] * sc + statesL[e];
    }
  }
}

// ------------- Y_off = (C @ state_in^T) * exp(Acum) ; add skip x*D -----------
__global__ void __launch_bounds__(128)
yoff_add(const float* __restrict__ xq, const float* __restrict__ Cm,
         const float* __restrict__ dt, const float* __restrict__ A_log,
         const float* __restrict__ statesIn, const float* __restrict__ Dp,
         float* __restrict__ y)
{
  const int h = blockIdx.x, c = blockIdx.y, b = blockIdx.z;
  const int tid = threadIdx.x, lane = tid & 31, wave = tid >> 5;
  __shared__ float sdt[64], sAc[64];
  __shared__ __align__(16) uint16_t Cl[64 * 64], Sl[64 * 64];
  const int tok0 = b * SEQLEN + c * 64;

  if (tid < 64) sdt[tid] = dt[(size_t)(tok0 + tid) * NHEADS + h];
  __syncthreads();
  if (tid == 0) {
    float Aval = -expf(A_log[h]), run = 0.f;
    for (int l = 0; l < 64; ++l) { run += Aval * sdt[l]; sAc[l] = run; }
  }
  __syncthreads();
  size_t sbase = ((size_t)((b * NCH + c) * NHEADS + h)) * 4096;
  for (int idx = tid; idx < 4096; idx += 128) {
    int r = idx >> 6, n = idx & 63;
    Cl[idx] = f2bf(Cm[(size_t)(tok0 + r) * DSTATE + n]);
    Sl[idx] = f2bf(statesIn[sbase + idx]);   // Sl[p][n]
  }
  __syncthreads();

  v8f acc[2][2] = {};
  gemm64(Cl, Sl, lane, wave, acc);
  const float Dv = Dp[h];
  const int rowB = (wave >> 1) * 32, colB = (wave & 1) * 32;
  const int mr = lane & 15, hi = lane >> 4;
#pragma unroll
  for (int i = 0; i < 2; ++i) {
    int rb = rowB + i * 16 + hi * 8;
#pragma unroll
    for (int j = 0; j < 2; ++j) {
      int pp = colB + j * 16 + mr;
#pragma unroll
      for (int r = 0; r < 8; ++r) {
        int l = rb + r;
        size_t yi = ((size_t)(tok0 + l)) * DINNER + h * HEADDIM + pp;
        y[yi] += acc[i][j][r] * expf(sAc[l]) + xq[yi] * Dv;
      }
    }
  }
}

// -------------------- SiLU gate + RMSNorm -> bf16 ----------------------------
__global__ void __launch_bounds__(256)
gate_rmsnorm(const float* __restrict__ zxbcdt, const float* __restrict__ y,
             const float* __restrict__ norm_w, uint16_t* __restrict__ ynbf)
{
  const int tok = blockIdx.x, tid = threadIdx.x;
  __shared__ float red[256];
  float v[8], ss = 0.f;
#pragma unroll
  for (int j = 0; j < 8; ++j) {
    int ch = tid + j * 256;
    float z = zxbcdt[(size_t)tok * DINPROJ + ch];
    float g = y[(size_t)tok * DINNER + ch] * (z * sigm(z));
    v[j] = g; ss += g * g;
  }
  red[tid] = ss; __syncthreads();
  for (int s = 128; s > 0; s >>= 1) {
    if (tid < s) red[tid] += red[tid + s];
    __syncthreads();
  }
  float scale = rsqrtf(red[0] / (float)DINNER + EPSF);
#pragma unroll
  for (int j = 0; j < 8; ++j) {
    int ch = tid + j * 256;
    ynbf[(size_t)tok * DINNER + ch] = f2bf(v[j] * scale * norm_w[ch]);
  }
}

// ---------------------------------------------------------------------------
extern "C" void kernel_launch(void* const* d_in, const int* in_sizes, int n_in,
                              void* d_out, int out_size, void* d_ws, size_t ws_size,
                              hipStream_t stream)
{
  (void)in_sizes; (void)n_in; (void)out_size; (void)ws_size;
  const float* u       = (const float*)d_in[0];
  const float* W_in    = (const float*)d_in[1];
  const float* conv_w  = (const float*)d_in[2];
  const float* conv_b  = (const float*)d_in[3];
  const float* dt_bias = (const float*)d_in[4];
  const float* A_log   = (const float*)d_in[5];
  const float* Dp      = (const float*)d_in[6];
  const float* norm_w  = (const float*)d_in[7];
  const float* W_out   = (const float*)d_in[8];
  float* out = (float*)d_out;

  char* ws = (char*)d_ws;
  size_t off = 0;
  auto alloc = [&](size_t bytes) -> void* {
    void* p = ws + off; off += (bytes + 255) & ~(size_t)255; return p;
  };
  float*    zxbcdt   = (float*)alloc((size_t)NTOK * DINPROJ * 4);
  float*    xq       = (float*)alloc((size_t)NTOK * DINNER * 4);
  float*    Bmw      = (float*)alloc((size_t)NTOK * DSTATE * 4);
  float*    Cmw      = (float*)alloc((size_t)NTOK * DSTATE * 4);
  float*    dtw      = (float*)alloc((size_t)NTOK * NHEADS * 4);
  float*    yw       = (float*)alloc((size_t)NTOK * DINNER * 4);
  float*    statesL  = (float*)alloc((size_t)BSZ * NCH * NHEADS * 4096 * 4);
  float*    statesIn = (float*)alloc((size_t)BSZ * NCH * NHEADS * 4096 * 4);
  float*    Acl      = (float*)alloc((size_t)BSZ * NCH * NHEADS * 4);
  uint16_t* ynbf     = (uint16_t*)alloc((size_t)NTOK * DINNER * 2);
  uint16_t* ubf      = (uint16_t*)alloc((size_t)NTOK * DMODEL * 2);
  uint16_t* winbf    = (uint16_t*)alloc((size_t)DINPROJ * DMODEL * 2);
  uint16_t* woutbf   = (uint16_t*)alloc((size_t)DMODEL * DINNER * 2);

  // one-shot operand conversions (bandwidth-bound, amortized across all tiles)
  {
    int n4;
    n4 = NTOK * DMODEL / 4;
    cvt_bf16x4<<<(n4 + 255) / 256, 256, 0, stream>>>(u, ubf, n4);
    n4 = DINPROJ * DMODEL / 4;
    cvt_bf16x4<<<(n4 + 255) / 256, 256, 0, stream>>>(W_in, winbf, n4);
    n4 = DMODEL * DINNER / 4;
    cvt_bf16x4<<<(n4 + 255) / 256, 256, 0, stream>>>(W_out, woutbf, n4);
  }

  gemm_nt_bb<<<dim3((DINPROJ + 63) / 64, NTOK / 128), 128, 0, stream>>>(
      ubf, winbf, zxbcdt, NTOK, DINPROJ, DMODEL);
  dt_softplus<<<(NTOK * NHEADS + 255) / 256, 256, 0, stream>>>(zxbcdt, dt_bias, dtw);
  conv_silu<<<NTOK, 256, 0, stream>>>(zxbcdt, conv_w, conv_b, xq, Bmw, Cmw);
  ssm_chunk<<<dim3(NHEADS, NCH, BSZ), 128, 0, stream>>>(
      xq, Bmw, Cmw, dtw, A_log, yw, statesL, Acl);
  state_scan<<<dim3(NHEADS, BSZ), 256, 0, stream>>>(statesL, Acl, statesIn);
  yoff_add<<<dim3(NHEADS, NCH, BSZ), 128, 0, stream>>>(
      xq, Cmw, dtw, A_log, statesIn, Dp, yw);
  gate_rmsnorm<<<NTOK, 256, 0, stream>>>(zxbcdt, yw, norm_w, ynbf);
  gemm_nt_bb<<<dim3(DMODEL / 64, NTOK / 128), 128, 0, stream>>>(
      ynbf, woutbf, out, NTOK, DMODEL, DINNER);
}
